// RNN_59055800320796
// MI455X (gfx1250) — matrix operand order repaired and checked by reading, not compiled, not run on hardware
//
#include <hip/hip_runtime.h>
#include <hip/hip_bf16.h>

typedef __attribute__((ext_vector_type(16))) _Float16 v16h;
typedef __attribute__((ext_vector_type(8)))  float    v8f;

#define TT   512
#define BB   512
#define KK   8
#define VV   5
#define HH   30
#define NLAY 4
#define G4   120   // 4*HH
#define NLAB 3

// Fast transcendental helpers: single v_exp_f32 + v_rcp_f32, no IEEE divide.
__device__ __forceinline__ float fast_sig(float x) {
    return __builtin_amdgcn_rcpf(1.0f + __expf(-x));
}
__device__ __forceinline__ float fast_tanh(float x) {
    // tanh(x) = 1 - 2/(exp(2x)+1); saturates correctly via exp->inf / exp->0
    return 1.0f - 2.0f * __builtin_amdgcn_rcpf(1.0f + __expf(2.0f * x));
}

// ---------------------------------------------------------------------------
// Kernel 1: embedding gather + weighted relu + mean over K  -> seq (T,B,H)
// ---------------------------------------------------------------------------
__global__ __launch_bounds__(256)
void embed_pool_kernel(const int* __restrict__ x, const float* __restrict__ wx,
                       const float* __restrict__ embed, float* __restrict__ seq)
{
    __shared__ float eL[VV * HH];
    const int tid = threadIdx.x;
    if (tid < VV * HH) eL[tid] = embed[tid];
    __syncthreads();

    const int tb = blockIdx.x * 256 + tid;
    if (tb >= TT * BB) return;

    float acc[HH];
#pragma unroll
    for (int h = 0; h < HH; ++h) acc[h] = 0.f;

    const size_t base = (size_t)tb * KK;
#pragma unroll
    for (int k = 0; k < KK; ++k) {
        const int   idx = x[base + k];
        const float wv  = wx[base + k];
        const float* er = &eL[idx * HH];
#pragma unroll
        for (int h = 0; h < HH; ++h) acc[h] += fmaxf(er[h] * wv, 0.f);
    }
    float* o = &seq[(size_t)tb * HH];
#pragma unroll
    for (int h = 0; h < HH; ++h) o[h] = acc[h] * (1.0f / KK);
}

// ---------------------------------------------------------------------------
// Kernel 2: 4-layer LSTM. One block owns 16 batch rows for ALL layers/steps.
// 8 waves: wave w computes gate columns [16w, 16w+16) via WMMA f32<-f16.
// Weights live in per-wave B-fragments (VGPRs); h/c live in LDS.
// ---------------------------------------------------------------------------
__global__ __launch_bounds__(256)
void lstm4_kernel(const float* __restrict__ Wih, const float* __restrict__ Whh,
                  const float* __restrict__ bih, const float* __restrict__ bhh,
                  float* __restrict__ bufA, float* __restrict__ bufB)
{
    __shared__ _Float16 xT[16 * 32];   // x_t tile, f16, K padded to 32
    __shared__ _Float16 hT[16 * 32];   // h_{t-1} tile, f16
    __shared__ float    cB[16 * 32];   // cell state, f32 (stride-32 padded)
    __shared__ float    gt[16 * 128];  // gate pre-activations, f32

    const int tid  = threadIdx.x;
    const int lane = tid & 31;
    const int wv   = tid >> 5;        // wave 0..7 -> N-tile
    const int nl   = lane & 15;
    const int hl   = lane >> 4;       // half-wave
    const int col  = wv * 16 + nl;    // gate column 0..127 (120 valid)
    const int row0 = blockIdx.x * 16; // batch row base

    for (int l = 0; l < NLAY; ++l) {
        const float* in  = (l & 1) ? bufB : bufA;
        float*       out = (l & 1) ? bufA : bufB;

        // Resident B-fragments: B[k][col] = W[col][k] (W is (4H,H) row-major)
        const float* Wi = Wih + (size_t)l * G4 * HH;
        const float* Wh = Whh + (size_t)l * G4 * HH;
        v16h bIH, bHH;
#pragma unroll
        for (int e = 0; e < 16; ++e) {
            const int k = (e < 8) ? (e + 8 * hl) : (8 + e + 8 * hl);
            float vi = 0.f, vh = 0.f;
            if (k < HH && col < G4) { vi = Wi[col * HH + k]; vh = Wh[col * HH + k]; }
            bIH[e] = (_Float16)vi;
            bHH[e] = (_Float16)vh;
        }
        const float bias = (col < G4) ? (bih[l * G4 + col] + bhh[l * G4 + col]) : 0.f;

        // zero state + padded tiles
        for (int i = tid; i < 16 * 32; i += 256) {
            xT[i] = (_Float16)0.f; hT[i] = (_Float16)0.f; cB[i] = 0.f;
        }
        __syncthreads();

        const float* inRow  = in  + (size_t)row0 * HH;
        float*       outRow = out + (size_t)row0 * HH;

        for (int t = 0; t < TT; ++t) {
            const size_t tOff = (size_t)t * BB * HH;
            // stage x_t tile (f32 -> f16); shift/mask indexing only
            for (int i = tid; i < 16 * 32; i += 256) {
                const int m = i >> 5, c = i & 31;
                if (c < HH)
                    xT[i] = (_Float16)inRow[tOff + (size_t)m * HH + c];
            }
            if (t + 1 < TT)
                __builtin_prefetch(&inRow[tOff + (size_t)BB * HH], 0, 3);
            __syncthreads();

            // A-fragments (ISA 16-bit 16x32 layout)
            v16h aX, aH;
#pragma unroll
            for (int j = 0; j < 8; ++j) {
                const int k0 = (j < 4) ? (2 * j + 8 * hl) : (16 + 2 * (j - 4) + 8 * hl);
                const int p  = nl * 32 + k0;
                aX[2 * j] = xT[p];  aX[2 * j + 1] = xT[p + 1];
                aH[2 * j] = hT[p];  aH[2 * j + 1] = hT[p + 1];
            }

            v8f d = {0.f, 0.f, 0.f, 0.f, 0.f, 0.f, 0.f, 0.f};
            d = __builtin_amdgcn_wmma_f32_16x16x32_f16(false, aX, false, bIH, (short)0, d, false, false);
            d = __builtin_amdgcn_wmma_f32_16x16x32_f16(false, aH, false, bHH, (short)0, d, false, false);

#pragma unroll
            for (int j = 0; j < 8; ++j) gt[(j + 8 * hl) * 128 + col] = d[j] + bias;
            __syncthreads();

            // elementwise LSTM cell (f32), rcp-based activations
            for (int i = tid; i < 16 * 32; i += 256) {
                const int m = i >> 5, hd = i & 31;
                if (hd < HH) {
                    const int g0 = m * 128 + hd;
                    const float gi = gt[g0];
                    const float gf = gt[g0 + HH];
                    const float gg = gt[g0 + 2 * HH];
                    const float go = gt[g0 + 3 * HH];
                    const float c  = fast_sig(gf) * cB[i] + fast_sig(gi) * fast_tanh(gg);
                    cB[i] = c;
                    const float h = fast_sig(go) * fast_tanh(c);
                    hT[i] = (_Float16)h;
                    outRow[tOff + (size_t)m * HH + hd] = h;
                }
            }
            __syncthreads();
        }
        __syncthreads();
    }
}

// ---------------------------------------------------------------------------
// Kernel 3: attention energy: relu((T*B,30) @ W1(30,64) + b1) @ W2 + b2
// M-tile = 16 contiguous rows of flattened (T,B); 4 waves cover N=64 via WMMA.
// ---------------------------------------------------------------------------
__global__ __launch_bounds__(128)
void energy_kernel(const float* __restrict__ hbuf, const float* __restrict__ W1,
                   const float* __restrict__ b1, const float* __restrict__ W2,
                   const float* __restrict__ b2, float* __restrict__ energy)
{
    __shared__ _Float16 aT[16 * 32];
    __shared__ float    part[16 * 64];

    const int tid  = threadIdx.x;
    const int lane = tid & 31;
    const int wv   = tid >> 5;       // 0..3 -> N-tile
    const int nl   = lane & 15;
    const int hl   = lane >> 4;
    const int n    = wv * 16 + nl;   // 0..63
    const size_t row0 = (size_t)blockIdx.x * 16;

    for (int i = tid; i < 16 * 32; i += 128) {
        const int m = i >> 5, c = i & 31;
        aT[i] = (c < HH) ? (_Float16)hbuf[(row0 + m) * HH + c] : (_Float16)0.f;
    }
    v16h bF;
#pragma unroll
    for (int e = 0; e < 16; ++e) {
        const int k = (e < 8) ? (e + 8 * hl) : (8 + e + 8 * hl);
        bF[e] = (_Float16)((k < HH) ? W1[k * 64 + n] : 0.f);
    }
    __syncthreads();

    v16h aF;
#pragma unroll
    for (int j = 0; j < 8; ++j) {
        const int k0 = (j < 4) ? (2 * j + 8 * hl) : (16 + 2 * (j - 4) + 8 * hl);
        const int p  = nl * 32 + k0;
        aF[2 * j] = aT[p]; aF[2 * j + 1] = aT[p + 1];
    }
    v8f d = {0.f, 0.f, 0.f, 0.f, 0.f, 0.f, 0.f, 0.f};
    d = __builtin_amdgcn_wmma_f32_16x16x32_f16(false, aF, false, bF, (short)0, d, false, false);

    const float b1n = b1[n], w2n = W2[n];
#pragma unroll
    for (int j = 0; j < 8; ++j)
        part[(j + 8 * hl) * 64 + n] = fmaxf(d[j] + b1n, 0.f) * w2n;
    __syncthreads();

    if (tid < 16) {
        float s = b2[0];
        for (int q = 0; q < 64; ++q) s += part[tid * 64 + q];
        energy[row0 + tid] = s;   // row = t*B + b
    }
}

// ---------------------------------------------------------------------------
// Kernel 4: per-batch softmax over T, weighted pooling, FC + softmax(3)
// ---------------------------------------------------------------------------
__global__ __launch_bounds__(256)
void pool_fc_kernel(const float* __restrict__ hbuf, const float* __restrict__ energy,
                    const float* __restrict__ fcW, const float* __restrict__ fcb,
                    float* __restrict__ outp)
{
    __shared__ float red[256];
    __shared__ float wb[TT];
    __shared__ float pool[32 * 8];
    __shared__ float pooled[32];

    const int tid = threadIdx.x;
    const int b   = blockIdx.x;

    float lm = -1e30f;
    for (int t = tid; t < TT; t += 256) lm = fmaxf(lm, energy[(size_t)t * BB + b]);
    red[tid] = lm; __syncthreads();
    for (int s = 128; s > 0; s >>= 1) { if (tid < s) red[tid] = fmaxf(red[tid], red[tid + s]); __syncthreads(); }
    const float mx = red[0]; __syncthreads();

    float ls = 0.f;
    for (int t = tid; t < TT; t += 256) {
        const float e = __expf(energy[(size_t)t * BB + b] - mx);
        wb[t] = e; ls += e;
    }
    red[tid] = ls; __syncthreads();
    for (int s = 128; s > 0; s >>= 1) { if (tid < s) red[tid] += red[tid + s]; __syncthreads(); }
    const float inv = __builtin_amdgcn_rcpf(red[0]); __syncthreads();

    const int h = tid & 31, ch = tid >> 5;   // 8 time-chunks of 64
    float s = 0.f;
    if (h < HH) {
        for (int t = ch * 64; t < ch * 64 + 64; ++t)
            s += wb[t] * hbuf[((size_t)t * BB + b) * HH + h];
    }
    pool[h * 8 + ch] = s;
    __syncthreads();
    if (tid < 32) {
        float p = 0.f;
#pragma unroll
        for (int c = 0; c < 8; ++c) p += pool[tid * 8 + c];
        pooled[tid] = p * inv;
    }
    __syncthreads();
    if (tid == 0) {
        float z[NLAB]; float zm = -1e30f;
        for (int n2 = 0; n2 < NLAB; ++n2) {
            float acc = fcb[n2];
            for (int hh = 0; hh < HH; ++hh) acc += pooled[hh] * fcW[hh * NLAB + n2];
            z[n2] = acc; zm = fmaxf(zm, acc);
        }
        float den = 0.f;
        for (int n2 = 0; n2 < NLAB; ++n2) { z[n2] = __expf(z[n2] - zm); den += z[n2]; }
        const float rd = 1.0f / den;
        for (int n2 = 0; n2 < NLAB; ++n2) outp[b * NLAB + n2] = z[n2] * rd;
    }
}

// ---------------------------------------------------------------------------
extern "C" void kernel_launch(void* const* d_in, const int* in_sizes, int n_in,
                              void* d_out, int out_size, void* d_ws, size_t ws_size,
                              hipStream_t stream)
{
    const int*   x   = (const int*)  d_in[0];
    const float* wx  = (const float*)d_in[1];
    const float* emb = (const float*)d_in[2];
    const float* Wih = (const float*)d_in[3];
    const float* Whh = (const float*)d_in[4];
    const float* bih = (const float*)d_in[5];
    const float* bhh = (const float*)d_in[6];
    const float* W1  = (const float*)d_in[7];
    const float* b1  = (const float*)d_in[8];
    const float* W2  = (const float*)d_in[9];
    const float* b2  = (const float*)d_in[10];
    const float* fcW = (const float*)d_in[11];
    const float* fcb = (const float*)d_in[12];
    float* out = (float*)d_out;

    const size_t TBH = (size_t)TT * BB * HH;
    float* bufA   = (float*)d_ws;         // seq / even-layer output / final h
    float* bufB   = bufA + TBH;           // odd-layer output
    float* energy = bufB + TBH;           // (T*B) attention energies

    embed_pool_kernel<<<(TT * BB) / 256, 256, 0, stream>>>(x, wx, emb, bufA);
    lstm4_kernel<<<BB / 16, 256, 0, stream>>>(Wih, Whh, bih, bhh, bufA, bufB);
    energy_kernel<<<(TT * BB) / 16, 128, 0, stream>>>(bufA, W1, b1, W2, b2, energy);
    pool_fc_kernel<<<BB, 256, 0, stream>>>(bufA, energy, fcW, fcb, out);
}
